// RoleAdapters_33947421507821
// MI455X (gfx1250) — matrix operand order repaired and checked by reading, not compiled, not run on hardware
//
#include <hip/hip_runtime.h>
#include <hip/hip_bf16.h>
#include <math.h>

typedef __attribute__((ext_vector_type(16))) __bf16 v16bf;
typedef __attribute__((ext_vector_type(8)))  float  v8f;
typedef __attribute__((ext_vector_type(4)))  unsigned int v4u;
typedef int v4i_vs __attribute__((vector_size(4 * sizeof(int))));

#define H_DIM 2048
#define K_DIM 1024
#define S_LEN 2048
#define B_SZ  8
#define LN_EPS 1e-5f

#if __has_builtin(__builtin_amdgcn_global_load_async_to_lds_b128)
#define USE_ASYNC_LDS 1
#else
#define USE_ASYNC_LDS 0
#endif

// ---- helpers -------------------------------------------------------------

__device__ __forceinline__ unsigned int f2bf_bits(float f) {
  unsigned int u = __float_as_uint(f);
  return (u + 0x7FFFu + ((u >> 16) & 1u)) >> 16;   // round-to-nearest-even
}
__device__ __forceinline__ unsigned int pack_bf16(float lo, float hi) {
  return (f2bf_bits(hi) << 16) | (f2bf_bits(lo) & 0xFFFFu);
}
__device__ __forceinline__ unsigned short f2bf(float f) {
  return (unsigned short)f2bf_bits(f);
}

// 16-byte global -> LDS copy (per lane). Async on gfx1250 when available.
__device__ __forceinline__ void cp16_g2l(const unsigned short* g, unsigned int* l) {
#if USE_ASYNC_LDS
  __builtin_amdgcn_global_load_async_to_lds_b128((v4i_vs*)g, (v4i_vs*)l, 0, 0);
#else
  *(v4u*)l = *(const v4u*)g;
#endif
}

template<int N>
__device__ __forceinline__ void wait_async() {
#if USE_ASYNC_LDS
#if __has_builtin(__builtin_amdgcn_s_wait_asynccnt)
  __builtin_amdgcn_s_wait_asynccnt(N);
#else
  asm volatile("s_wait_asynccnt %0" :: "i"(N));
#endif
#endif
}

union AFrag { v16bf v; unsigned int u[8]; };
union BFrag { v16bf v; unsigned int u[8]; };

// ---- bf16 WMMA GEMM: C[m][n] = sum_k A[m][k] * Bt[n][k] ------------------
// Block 256 threads (8 wave32), block tile 128x128, K-step 32, double-buffered
// LDS staged via GLOBAL_LOAD_ASYNC_TO_LDS_B128. Row stride 20 dwords (80 B,
// 16B-aligned rows; fragment b128 reads bank-conflict-free).
//
// GELU_OUT==true : O = GELU(C + bias) -> bf16
// GELU_OUT==false: O = C + bias + X   -> f32

template<bool GELU_OUT, int N_DIM, int KD>
__global__ __launch_bounds__(256)
void gemm_bf16(const unsigned short* __restrict__ Aall,  // [B, S, KD] bf16
               const unsigned short* __restrict__ Btall, // [R, N_DIM, KD] bf16 (pre-transposed)
               const float* __restrict__ ball,           // [R, N_DIM]
               const float* __restrict__ X,              // [B, S, N_DIM] residual
               unsigned short* __restrict__ Obf,         // bf16 out (GELU path)
               float* __restrict__ Of32,                 // f32 out (residual path)
               const int* __restrict__ role)
{
  const int b      = blockIdx.z;
  const int r      = role[b];
  const int mBlock = blockIdx.y * 128;
  const int nBlock = blockIdx.x * 128;

  const unsigned short* Ab = Aall  + (size_t)b * S_LEN * KD + (size_t)mBlock * KD;
  const unsigned short* Bb = Btall + (size_t)r * N_DIM * KD + (size_t)nBlock * KD;
  const float* bias = ball + (size_t)r * N_DIM;

  __shared__ unsigned int lA[2][128][20];
  __shared__ unsigned int lB[2][128][20];

  const int tid   = threadIdx.x;
  const int lane  = tid & 31;
  const int wave  = tid >> 5;
  const int wm    = (wave & 3) * 32;
  const int wn    = (wave >> 2) * 64;
  const int lmod  = lane & 15;
  const int lhalf = lane >> 4;

  // Stage both 128x32 bf16 tiles: 1024 16B chunks = 32 wave-instructions,
  // 4 per wave (waves 0-3 load A, waves 4-7 load B).
  auto stage = [&](int kt, int buf) {
#pragma unroll
    for (int t = 0; t < 4; ++t) {
      const int inst = wave * 4 + t;
      const int ci   = (inst & 15) * 32 + lane;   // chunk 0..511
      const int row  = ci >> 2;
      const int q    = ci & 3;
      if (inst < 16)
        cp16_g2l(Ab + (size_t)row * KD + kt * 32 + q * 8, &lA[buf][row][q * 4]);
      else
        cp16_g2l(Bb + (size_t)row * KD + kt * 32 + q * 8, &lB[buf][row][q * 4]);
    }
  };

  const v8f vzero = {0.f,0.f,0.f,0.f,0.f,0.f,0.f,0.f};
  v8f acc[2][4];
#pragma unroll
  for (int mi = 0; mi < 2; ++mi)
#pragma unroll
    for (int ni = 0; ni < 4; ++ni) acc[mi][ni] = vzero;

  constexpr int KT = KD / 32;
  stage(0, 0);

  for (int kt = 0; kt < KT; ++kt) {
    if (kt + 1 < KT) {                 // prefetch next stage into other buffer
      stage(kt + 1, (kt + 1) & 1);
      wait_async<4>();                 // own next-stage still in flight; prev done
    } else {
      wait_async<0>();
    }
    __syncthreads();

    const unsigned int (*cA)[20] = lA[kt & 1];
    const unsigned int (*cB)[20] = lB[kt & 1];

    AFrag a[2];
#pragma unroll
    for (int mi = 0; mi < 2; ++mi) {
      const unsigned int* src = &cA[wm + mi * 16 + lmod][0];
#pragma unroll
      for (int j = 0; j < 4; ++j) a[mi].u[j]     = src[lhalf * 4 + j];      // K 0..7 / 8..15
#pragma unroll
      for (int j = 0; j < 4; ++j) a[mi].u[4 + j] = src[8 + lhalf * 4 + j];  // K 16..23 / 24..31
    }
    BFrag bb[4];
#pragma unroll
    for (int ni = 0; ni < 4; ++ni) {
      const unsigned int* src = &cB[wn + ni * 16 + lmod][0];
#pragma unroll
      for (int j = 0; j < 4; ++j) bb[ni].u[j]     = src[lhalf * 4 + j];      // K 0..7 / 8..15
#pragma unroll
      for (int j = 0; j < 4; ++j) bb[ni].u[4 + j] = src[8 + lhalf * 4 + j];  // K 16..23 / 24..31
    }
#pragma unroll
    for (int mi = 0; mi < 2; ++mi)
#pragma unroll
      for (int ni = 0; ni < 4; ++ni)
        acc[mi][ni] = __builtin_amdgcn_wmma_f32_16x16x32_bf16(
            false, a[mi].v, false, bb[ni].v, (short)0, acc[mi][ni], false, false);

    __syncthreads();
  }

  // ---- epilogue ----
  unsigned short* Ob = GELU_OUT ? (Obf  + (size_t)b * S_LEN * N_DIM) : nullptr;
  float*          Oy = GELU_OUT ? nullptr : (Of32 + (size_t)b * S_LEN * N_DIM);
  const float*    Xb = GELU_OUT ? nullptr : (X    + (size_t)b * S_LEN * N_DIM);

#pragma unroll
  for (int mi = 0; mi < 2; ++mi) {
#pragma unroll
    for (int ni = 0; ni < 4; ++ni) {
      const int ncol = nBlock + wn + ni * 16 + lmod;
      const float bs = bias[ncol];
#pragma unroll
      for (int rr = 0; rr < 8; ++rr) {
        const int mrow = mBlock + wm + mi * 16 + lhalf * 8 + rr;  // lanes 16-31 -> M+8
        float v = acc[mi][ni][rr] + bs;
        if constexpr (GELU_OUT) {
          v = 0.5f * v * (1.0f + erff(v * 0.70710678118654752f)); // exact-erf GELU
          Ob[(size_t)mrow * N_DIM + ncol] = f2bf(v);
        } else {
          v += Xb[(size_t)mrow * N_DIM + ncol];
          Oy[(size_t)mrow * N_DIM + ncol] = v;
        }
      }
    }
  }
}

// ---- f32 -> bf16 bulk convert (vectorized) -------------------------------

__global__ __launch_bounds__(256)
void cvt_bf16(const float* __restrict__ in, unsigned short* __restrict__ out, int n4)
{
  const int i = blockIdx.x * 256 + threadIdx.x;
  if (i < n4) {
    const float4 v = ((const float4*)in)[i];
    uint2 o;
    o.x = pack_bf16(v.x, v.y);
    o.y = pack_bf16(v.z, v.w);
    ((uint2*)out)[i] = o;
  }
}

// ---- transpose + convert: out[mat][c][r] = (bf16)in[mat][r][c] -----------

__global__ __launch_bounds__(256)
void transpose_cvt(const float* __restrict__ in, unsigned short* __restrict__ out,
                   int R_, int C_)
{
  __shared__ float t[32][33];
  const int mat = blockIdx.z;
  const float*    ip = in  + (size_t)mat * R_ * C_;
  unsigned short* op = out + (size_t)mat * R_ * C_;
  const int c0 = blockIdx.x * 32, r0 = blockIdx.y * 32;
  const int tx = threadIdx.x & 31, ty = threadIdx.x >> 5;   // 32 x 8
#pragma unroll
  for (int i = 0; i < 32; i += 8)
    t[ty + i][tx] = ip[(size_t)(r0 + ty + i) * C_ + c0 + tx];
  __syncthreads();
#pragma unroll
  for (int i = 0; i < 32; i += 8)
    op[(size_t)(c0 + ty + i) * R_ + r0 + tx] = f2bf(t[tx][ty + i]);
}

// ---- LayerNorm over H with per-role affine (vectorized) ------------------

__global__ __launch_bounds__(256)
void ln_kernel(const float* __restrict__ Y, const float* __restrict__ gamma,
               const float* __restrict__ beta, const int* __restrict__ role,
               float* __restrict__ out)
{
  const int row = blockIdx.x;            // 0 .. B*S-1
  const int b   = row >> 11;             // row / S_LEN
  const int r   = role[b];
  const float4* y4  = (const float4*)(Y   + (size_t)row * H_DIM);
  float4*       o4  = (float4*)(out       + (size_t)row * H_DIM);
  const float4* g4  = (const float4*)(gamma + (size_t)r * H_DIM);
  const float4* be4 = (const float4*)(beta  + (size_t)r * H_DIM);

  const int tid = threadIdx.x;
  float4 v0 = y4[tid], v1 = y4[tid + 256];
  float s  = v0.x + v0.y + v0.z + v0.w + v1.x + v1.y + v1.z + v1.w;
  float ss = v0.x*v0.x + v0.y*v0.y + v0.z*v0.z + v0.w*v0.w
           + v1.x*v1.x + v1.y*v1.y + v1.z*v1.z + v1.w*v1.w;

  __shared__ float rs[256], rss[256];
  rs[tid] = s; rss[tid] = ss;
  __syncthreads();
  for (int off = 128; off > 0; off >>= 1) {
    if (tid < off) { rs[tid] += rs[tid + off]; rss[tid] += rss[tid + off]; }
    __syncthreads();
  }
  const float mean = rs[0] * (1.0f / H_DIM);
  const float var  = rss[0] * (1.0f / H_DIM) - mean * mean;
  const float inv  = rsqrtf(var + LN_EPS);

  float4 g0 = g4[tid], g1 = g4[tid + 256];
  float4 b0 = be4[tid], b1 = be4[tid + 256];
  float4 r0, r1;
  r0.x = (v0.x - mean) * inv * g0.x + b0.x;
  r0.y = (v0.y - mean) * inv * g0.y + b0.y;
  r0.z = (v0.z - mean) * inv * g0.z + b0.z;
  r0.w = (v0.w - mean) * inv * g0.w + b0.w;
  r1.x = (v1.x - mean) * inv * g1.x + b1.x;
  r1.y = (v1.y - mean) * inv * g1.y + b1.y;
  r1.z = (v1.z - mean) * inv * g1.z + b1.z;
  r1.w = (v1.w - mean) * inv * g1.w + b1.w;
  o4[tid] = r0;
  o4[tid + 256] = r1;
}

// ---- launch --------------------------------------------------------------

extern "C" void kernel_launch(void* const* d_in, const int* in_sizes, int n_in,
                              void* d_out, int out_size, void* d_ws, size_t ws_size,
                              hipStream_t stream) {
  const int*   role  = (const int*)  d_in[0];
  const float* hs    = (const float*)d_in[1];
  const float* Wd    = (const float*)d_in[2];  // [R, H, K]
  const float* bd    = (const float*)d_in[3];
  const float* Wu    = (const float*)d_in[4];  // [R, K, H]
  const float* bu    = (const float*)d_in[5];
  const float* gamma = (const float*)d_in[6];
  const float* beta  = (const float*)d_in[7];
  float* out = (float*)d_out;

  // workspace layout (bytes)
  char* ws = (char*)d_ws;
  unsigned short* xbf  = (unsigned short*)(ws);                        // B*S*H bf16  = 64 MiB
  unsigned short* h_ws = (unsigned short*)(ws + (size_t)67108864);     // B*S*K bf16  = 32 MiB
  unsigned short* Wdt  = (unsigned short*)(ws + (size_t)100663296);    // R*K*H bf16  = 12 MiB
  unsigned short* Wut  = (unsigned short*)(ws + (size_t)113246208);    // R*H*K bf16  = 12 MiB
  float*          y_ws = (float*)         (ws + (size_t)125829120);    // B*S*H f32   = 128 MiB

  dim3 blk(256);

  // 1) x -> bf16
  {
    const int n4 = (B_SZ * S_LEN * H_DIM) / 4;
    cvt_bf16<<<n4 / 256, blk, 0, stream>>>(hs, xbf, n4);
  }
  // 2) weights -> bf16, transposed to [N][K]
  transpose_cvt<<<dim3(K_DIM / 32, H_DIM / 32, 3), blk, 0, stream>>>(Wd, Wdt, H_DIM, K_DIM);
  transpose_cvt<<<dim3(H_DIM / 32, K_DIM / 32, 3), blk, 0, stream>>>(Wu, Wut, K_DIM, H_DIM);

  // 3) h = GELU(x @ W_down + b_down)          (bf16 out)
  gemm_bf16<true, K_DIM, H_DIM><<<dim3(K_DIM / 128, S_LEN / 128, B_SZ), blk, 0, stream>>>(
      xbf, Wdt, bd, nullptr, h_ws, nullptr, role);

  // 4) y = h @ W_up + b_up + x                (f32 out)
  gemm_bf16<false, H_DIM, K_DIM><<<dim3(H_DIM / 128, S_LEN / 128, B_SZ), blk, 0, stream>>>(
      h_ws, Wut, bu, hs, nullptr, y_ws, role);

  // 5) LayerNorm + per-role affine
  ln_kernel<<<B_SZ * S_LEN, blk, 0, stream>>>(y_ws, gamma, beta, role, out);
}
